// DeformableTransformer_70970039599649
// MI455X (gfx1250) — matrix operand order repaired and checked by reading, not compiled, not run on hardware
//
#include <hip/hip_runtime.h>
#include <hip/hip_bf16.h>
#include <stdint.h>

typedef _Float16 v8h  __attribute__((ext_vector_type(8)));
typedef _Float16 v16h __attribute__((ext_vector_type(16)));
typedef float    v8f  __attribute__((ext_vector_type(8)));

#define B_    8
#define LQ    900
#define D_    256
#define LIN   21760
#define MQ    (B_*LQ)      // 7200
#define MV    (B_*LIN)     // 174080
#define DFF_  1024
#define NH_   8
#define DH_   32

// ---------------------------------------------------------------------------
// elementwise helpers
// ---------------------------------------------------------------------------
__global__ void k_cast_f16(const float* __restrict__ a, _Float16* __restrict__ o, long long n) {
    long long i = (long long)blockIdx.x * 256 + threadIdx.x;
    if (i < n) o[i] = (_Float16)a[i];
}

__global__ void k_addcast_f16(const float* __restrict__ a, const float* __restrict__ b,
                              _Float16* __restrict__ o, long long n) {
    long long i = (long long)blockIdx.x * 256 + threadIdx.x;
    if (i < n) o[i] = (_Float16)(a[i] + b[i]);
}

// wt[n*K+k] = (f16) w[k*N+n]
__global__ void k_transpose_w(const float* __restrict__ w, _Float16* __restrict__ wt,
                              int K, int N) {
    long long i = (long long)blockIdx.x * 256 + threadIdx.x;
    if (i >= (long long)K * N) return;
    int n = (int)(i / K), k = (int)(i % K);
    wt[i] = (_Float16)w[(size_t)k * N + n];
}

__global__ void k_mask_value(_Float16* __restrict__ v, const unsigned char* __restrict__ mask,
                             long long n) {
    long long i = (long long)blockIdx.x * 256 + threadIdx.x;
    if (i < n && mask[i >> 8]) v[i] = (_Float16)0.f;
}

__global__ void k_softmax16(float* __restrict__ a, int cnt) {
    int i = blockIdx.x * 256 + threadIdx.x;
    if (i >= cnt) return;
    float* p = a + (size_t)i * 16;
    float m = -1e30f;
    #pragma unroll
    for (int j = 0; j < 16; j++) m = fmaxf(m, p[j]);
    float e[16], s = 0.f;
    #pragma unroll
    for (int j = 0; j < 16; j++) { e[j] = __expf(p[j] - m); s += e[j]; }
    float inv = 1.f / s;
    #pragma unroll
    for (int j = 0; j < 16; j++) p[j] = e[j] * inv;
}

// ---------------------------------------------------------------------------
// LayerNorm: y = LN(x + res) * g + beta ; optional f16 copy of (y + pos)
// ---------------------------------------------------------------------------
__global__ __launch_bounds__(256)
void k_layernorm(const float* __restrict__ x, const float* __restrict__ res,
                 const float* __restrict__ g, const float* __restrict__ beta,
                 const float* __restrict__ pos, float* __restrict__ outf,
                 _Float16* __restrict__ outh, int rows) {
    int row = blockIdx.x * 8 + (threadIdx.x >> 5);
    if (row >= rows) return;
    int lane = threadIdx.x & 31;
    const float* xp = x + (size_t)row * D_;
    const float* rp = res + (size_t)row * D_;
    float v[8], s = 0.f;
    #pragma unroll
    for (int i = 0; i < 8; i++) { v[i] = xp[lane + 32 * i] + rp[lane + 32 * i]; s += v[i]; }
    #pragma unroll
    for (int m = 1; m < 32; m <<= 1) s += __shfl_xor(s, m, 32);
    float mean = s * (1.f / 256.f), vs = 0.f;
    #pragma unroll
    for (int i = 0; i < 8; i++) { float d = v[i] - mean; vs += d * d; }
    #pragma unroll
    for (int m = 1; m < 32; m <<= 1) vs += __shfl_xor(vs, m, 32);
    float inv = rsqrtf(vs * (1.f / 256.f) + 1e-5f);
    #pragma unroll
    for (int i = 0; i < 8; i++) {
        int c = lane + 32 * i;
        float y = (v[i] - mean) * inv * g[c] + beta[c];
        size_t o = (size_t)row * D_ + c;
        if (outf) outf[o] = y;
        if (outh) outh[o] = (_Float16)(y + (pos ? pos[o] : 0.f));
    }
}

// ---------------------------------------------------------------------------
// Block-cooperative WMMA GEMM with async-to-LDS double buffering.
// out[M][N] = A[M][K] * Wt[N][K]^T + bias
// Block = 256 threads (8 waves); block tile 64(M) x 64(N); wave tile 16x32.
// Per k-step (K32): A tile 64x32 f16 + B tile 64x32 f16 staged via
// GLOBAL_LOAD_ASYNC_TO_LDS_B128 (ASYNCcnt), double buffered.
// ---------------------------------------------------------------------------
__device__ __forceinline__ void async_ld16(uint32_t lds, const _Float16* g) {
    asm volatile("global_load_async_to_lds_b128 %0, %1, off"
                 :: "v"(lds), "v"((uint64_t)(uintptr_t)g) : "memory");
}

__global__ __launch_bounds__(256)
void k_gemm(const _Float16* __restrict__ A, const _Float16* __restrict__ Wt,
            const float* __restrict__ bias, float* __restrict__ outf,
            _Float16* __restrict__ outh, int M, int N, int K, int relu) {
    __shared__ _Float16 As[2][64 * 32];
    __shared__ _Float16 Bs[2][64 * 32];
    int t = threadIdx.x;
    int w = t >> 5, lane = t & 31;
    int l16 = lane & 15, half = lane >> 4;
    int wm = w >> 1, wn = w & 1;           // 4 x 2 wave grid; wave = 16 x 32
    int m0 = blockIdx.y * 64, n0 = blockIdx.x * 64;

    // cooperative loader: thread t -> row t/4 (0..63), 16B chunk (t&3)*8 halves
    int lrow = t >> 2;
    int lch = (t & 3) * 8;
    int arow = m0 + lrow; if (arow > M - 1) arow = M - 1;
    const _Float16* agp = A + (size_t)arow * K + lch;
    const _Float16* bgp = Wt + (size_t)(n0 + lrow) * K + lch;
    uint32_t aLds[2], bLds[2];
    aLds[0] = (uint32_t)(uintptr_t)&As[0][lrow * 32 + lch];
    aLds[1] = (uint32_t)(uintptr_t)&As[1][lrow * 32 + lch];
    bLds[0] = (uint32_t)(uintptr_t)&Bs[0][lrow * 32 + lch];
    bLds[1] = (uint32_t)(uintptr_t)&Bs[1][lrow * 32 + lch];

    const int nk = K / 32;
    async_ld16(aLds[0], agp);
    async_ld16(bLds[0], bgp);

    v8f acc0 = {}, acc1 = {};
    for (int i = 0; i < nk; i++) {
        asm volatile("s_wait_asynccnt 0" ::: "memory");
        __syncthreads();
        if (i + 1 < nk) {
            int nb = (i + 1) & 1, k0 = (i + 1) * 32;
            async_ld16(aLds[nb], agp + k0);
            async_ld16(bLds[nb], bgp + k0);
        }
        const _Float16* At = As[i & 1];
        const _Float16* Bt = Bs[i & 1];
        v16h a;
        {
            v8h lo = *(const v8h*)&At[(wm * 16 + l16) * 32 + half * 16];
            v8h hi = *(const v8h*)&At[(wm * 16 + l16) * 32 + half * 16 + 8];
            #pragma unroll
            for (int j = 0; j < 8; j++) { a[j] = lo[j]; a[j + 8] = hi[j]; }
        }
        v16h b0 = *(const v16h*)&Bt[(wn * 32 + l16) * 32 + half * 16];
        v16h b1 = *(const v16h*)&Bt[(wn * 32 + 16 + l16) * 32 + half * 16];
        acc0 = __builtin_amdgcn_wmma_f32_16x16x32_f16(false, a, false, b0,
                                                      (short)0, acc0, false, false);
        acc1 = __builtin_amdgcn_wmma_f32_16x16x32_f16(false, a, false, b1,
                                                      (short)0, acc1, false, false);
    }

    int col0 = n0 + wn * 32 + l16;
    float bv0 = bias ? bias[col0] : 0.f;
    float bv1 = bias ? bias[col0 + 16] : 0.f;
    #pragma unroll
    for (int r = 0; r < 8; r++) {
        int row = m0 + wm * 16 + r + 8 * half;
        if (row >= M) continue;
        float v0 = acc0[r] + bv0, v1 = acc1[r] + bv1;
        if (relu) { v0 = fmaxf(v0, 0.f); v1 = fmaxf(v1, 0.f); }
        size_t o = (size_t)row * N + col0;
        if (outf) { outf[o] = v0; outf[o + 16] = v1; }
        if (outh) { outh[o] = (_Float16)v0; outh[o + 16] = (_Float16)v1; }
    }
}

// ---------------------------------------------------------------------------
// Flash self-attention: one wave per (b, h, 16-row Q tile), DH=32
// qh/kh/vh: f16 [B*LQ][256], head h occupies cols h*32..h*32+31
// ---------------------------------------------------------------------------
__global__ __launch_bounds__(32)
void k_flash(const _Float16* __restrict__ Q, const _Float16* __restrict__ Kh,
             const _Float16* __restrict__ V, _Float16* __restrict__ O) {
    __shared__ _Float16 P[16 * 32];
    int wid = blockIdx.x;                 // bh*57 + qt
    int qt = wid % 57;
    int bh = wid / 57;
    int h = bh % NH_, b = bh / NH_;
    int lane = threadIdx.x;
    int l16 = lane & 15, half = lane >> 4;
    const float scale = 0.17677669529663687f;   // 1/sqrt(32)

    int qrow = qt * 16 + l16; if (qrow > LQ - 1) qrow = LQ - 1;
    const _Float16* qp = Q + (size_t)(b * LQ + qrow) * D_ + h * DH_;
    int kb = half * 16;       // A-fragment K base
    int ks = half * 16;       // B-fragment K base
    v16h aq;
    {
        v8h lo = *(const v8h*)(qp + kb);
        v8h hi = *(const v8h*)(qp + kb + 8);
        #pragma unroll
        for (int i = 0; i < 8; i++) { aq[i] = lo[i]; aq[i + 8] = hi[i]; }
    }

    float mrow[8], lrow[8];
    v8f o0 = {}, o1 = {};
    #pragma unroll
    for (int r = 0; r < 8; r++) { mrow[r] = -1e30f; lrow[r] = 0.f; }

    for (int kt = 0; kt < LQ; kt += 32) {
        v8f s[2];
        #pragma unroll
        for (int j = 0; j < 2; j++) {
            int key = kt + j * 16 + l16; if (key > LQ - 1) key = LQ - 1;
            const _Float16* kp = Kh + (size_t)(b * LQ + key) * D_ + h * DH_ + ks;
            v16h bk = *(const v16h*)kp;
            v8f z = {};
            s[j] = __builtin_amdgcn_wmma_f32_16x16x32_f16(false, aq, false, bk,
                                                          (short)0, z, false, false);
        }
        #pragma unroll
        for (int j = 0; j < 2; j++) {
            bool bad = (kt + j * 16 + l16) >= LQ;
            #pragma unroll
            for (int r = 0; r < 8; r++) s[j][r] = bad ? -1e30f : s[j][r] * scale;
        }
        float nm[8], rs[8];
        #pragma unroll
        for (int r = 0; r < 8; r++) {
            float v = fmaxf(s[0][r], s[1][r]);
            #pragma unroll
            for (int m = 1; m < 16; m <<= 1) v = fmaxf(v, __shfl_xor(v, m, 32));
            nm[r] = fmaxf(v, mrow[r]);
        }
        #pragma unroll
        for (int r = 0; r < 8; r++) {
            float p0 = __expf(s[0][r] - nm[r]);
            float p1 = __expf(s[1][r] - nm[r]);
            s[0][r] = p0; s[1][r] = p1;
            float v = p0 + p1;
            #pragma unroll
            for (int m = 1; m < 16; m <<= 1) v += __shfl_xor(v, m, 32);
            rs[r] = v;
        }
        #pragma unroll
        for (int r = 0; r < 8; r++) {
            float al = __expf(mrow[r] - nm[r]);
            lrow[r] = lrow[r] * al + rs[r];
            mrow[r] = nm[r];
            o0[r] *= al; o1[r] *= al;
        }
        #pragma unroll
        for (int j = 0; j < 2; j++)
            #pragma unroll
            for (int r = 0; r < 8; r++)
                P[(r + 8 * half) * 32 + j * 16 + l16] = (_Float16)s[j][r];
        __syncthreads();
        v16h ap;
        {
            v8h lo = *(const v8h*)&P[l16 * 32 + kb];
            v8h hi = *(const v8h*)&P[l16 * 32 + kb + 8];
            #pragma unroll
            for (int i = 0; i < 8; i++) { ap[i] = lo[i]; ap[i + 8] = hi[i]; }
        }
        #pragma unroll
        for (int t = 0; t < 2; t++) {
            v16h bv;
            #pragma unroll
            for (int i = 0; i < 16; i++) {
                int key = kt + ks + i; if (key > LQ - 1) key = LQ - 1;
                bv[i] = V[(size_t)(b * LQ + key) * D_ + h * DH_ + t * 16 + l16];
            }
            if (t == 0)
                o0 = __builtin_amdgcn_wmma_f32_16x16x32_f16(false, ap, false, bv,
                                                            (short)0, o0, false, false);
            else
                o1 = __builtin_amdgcn_wmma_f32_16x16x32_f16(false, ap, false, bv,
                                                            (short)0, o1, false, false);
        }
        __syncthreads();
    }
    #pragma unroll
    for (int r = 0; r < 8; r++) {
        float inv = 1.f / lrow[r];
        int row = qt * 16 + r + 8 * half;
        if (row < LQ) {
            size_t base = (size_t)(b * LQ + row) * D_ + h * DH_ + l16;
            O[base]      = (_Float16)(o0[r] * inv);
            O[base + 16] = (_Float16)(o1[r] * inv);
        }
    }
}

// ---------------------------------------------------------------------------
// MS deformable attention gather: one wave per (b,q,h); lane = channel (DH=32)
// ---------------------------------------------------------------------------
__global__ __launch_bounds__(256)
void k_deform(const _Float16* __restrict__ value, const float* __restrict__ off,
              const float* __restrict__ aw, const float* __restrict__ refp,
              _Float16* __restrict__ out) {
    int wid = blockIdx.x * 8 + (threadIdx.x >> 5);
    int lane = threadIdx.x & 31;
    int h = wid & 7;
    int q = (wid >> 3) % LQ;
    int b = wid / (LQ * NH_);
    const int HS[4] = {128, 64, 32, 16};
    const int WS[4] = {128, 64, 32, 16};
    const int S0[4] = {0, 16384, 20480, 21504};
    const float* offp = off + (size_t)(b * LQ + q) * D_ + h * 32;
    const float* awp  = aw  + (size_t)(b * LQ + q) * 128 + h * 16;
    const float* rp   = refp + (size_t)(b * LQ + q) * 8;
    size_t vbase = (size_t)b * LIN * D_ + h * DH_ + lane;
    float acc = 0.f;
    #pragma unroll
    for (int l = 0; l < 4; l++) {
        int H = HS[l], W = WS[l], s0 = S0[l];
        float rx = rp[l * 2 + 0], ry = rp[l * 2 + 1];
        float invW = 1.f / (float)W, invH = 1.f / (float)H;
        #pragma unroll
        for (int p = 0; p < 4; p++) {
            float ox = offp[(l * 4 + p) * 2 + 0];
            float oy = offp[(l * 4 + p) * 2 + 1];
            float wgt = awp[l * 4 + p];
            float x = (rx + ox * invW) * (float)W - 0.5f;
            float y = (ry + oy * invH) * (float)H - 0.5f;
            float xf = floorf(x), yf = floorf(y);
            int x0 = (int)xf, y0 = (int)yf;
            float wx1 = x - xf, wy1 = y - yf;
            float wx0 = 1.f - wx1, wy0 = 1.f - wy1;
            float g = 0.f;
            #pragma unroll
            for (int cy = 0; cy < 2; cy++) {
                int yy = y0 + cy;
                float wy = cy ? wy1 : wy0;
                #pragma unroll
                for (int cx = 0; cx < 2; cx++) {
                    int xx = x0 + cx;
                    float wx = cx ? wx1 : wx0;
                    if (yy >= 0 && yy < H && xx >= 0 && xx < W) {
                        int idx = s0 + yy * W + xx;
                        g += wy * wx * (float)value[vbase + (size_t)idx * D_];
                    }
                }
            }
            acc += wgt * g;
        }
    }
    out[(size_t)(b * LQ + q) * D_ + h * DH_ + lane] = (_Float16)acc;
}

// ---------------------------------------------------------------------------
// launcher
// ---------------------------------------------------------------------------
extern "C" void kernel_launch(void* const* d_in, const int* in_sizes, int n_in,
                              void* d_out, int out_size, void* d_ws, size_t ws_size,
                              hipStream_t stream) {
    const float* tgt   = (const float*)d_in[0];
    const float* qpos  = (const float*)d_in[1];
    const float* refp  = (const float*)d_in[2];
    const float* src   = (const float*)d_in[3];
    const unsigned char* pmask = (const unsigned char*)d_in[6];
    const float* w_q  = (const float*)d_in[7],  *b_q  = (const float*)d_in[8];
    const float* w_k  = (const float*)d_in[9],  *b_k  = (const float*)d_in[10];
    const float* w_v  = (const float*)d_in[11], *b_v  = (const float*)d_in[12];
    const float* w_o  = (const float*)d_in[13], *b_o  = (const float*)d_in[14];
    const float* w_vp = (const float*)d_in[15], *b_vp = (const float*)d_in[16];
    const float* w_so = (const float*)d_in[17], *b_so = (const float*)d_in[18];
    const float* w_aw = (const float*)d_in[19], *b_aw = (const float*)d_in[20];
    const float* w_op = (const float*)d_in[21], *b_op = (const float*)d_in[22];
    const float* w_l1 = (const float*)d_in[23], *b_l1 = (const float*)d_in[24];
    const float* w_l2 = (const float*)d_in[25], *b_l2 = (const float*)d_in[26];
    const float* ln1g = (const float*)d_in[27], *ln1b = (const float*)d_in[28];
    const float* ln2g = (const float*)d_in[29], *ln2b = (const float*)d_in[30];
    const float* ln3g = (const float*)d_in[31], *ln3b = (const float*)d_in[32];

    char* ws = (char*)d_ws;
    size_t off = 0;
    auto alloc = [&](size_t bytes) -> void* {
        void* p = ws + off;
        off = (off + bytes + 255) & ~(size_t)255;
        return p;
    };
    _Float16* wt_q  = (_Float16*)alloc((size_t)256 * 256 * 2);
    _Float16* wt_k  = (_Float16*)alloc((size_t)256 * 256 * 2);
    _Float16* wt_v  = (_Float16*)alloc((size_t)256 * 256 * 2);
    _Float16* wt_o  = (_Float16*)alloc((size_t)256 * 256 * 2);
    _Float16* wt_vp = (_Float16*)alloc((size_t)256 * 256 * 2);
    _Float16* wt_so = (_Float16*)alloc((size_t)256 * 256 * 2);
    _Float16* wt_aw = (_Float16*)alloc((size_t)256 * 128 * 2);
    _Float16* wt_op = (_Float16*)alloc((size_t)256 * 256 * 2);
    _Float16* wt_l1 = (_Float16*)alloc((size_t)256 * 1024 * 2);
    _Float16* wt_l2 = (_Float16*)alloc((size_t)1024 * 256 * 2);
    _Float16* qin_h = (_Float16*)alloc((size_t)MQ * D_ * 2);
    _Float16* tgt_h = (_Float16*)alloc((size_t)MQ * D_ * 2);
    _Float16* qh    = (_Float16*)alloc((size_t)MQ * D_ * 2);
    _Float16* kh    = (_Float16*)alloc((size_t)MQ * D_ * 2);
    _Float16* vh    = (_Float16*)alloc((size_t)MQ * D_ * 2);
    _Float16* sa_h  = (_Float16*)alloc((size_t)MQ * D_ * 2);
    float*    attn_f= (float*)   alloc((size_t)MQ * D_ * 4);
    float*    tgt1_f= (float*)   alloc((size_t)MQ * D_ * 4);
    _Float16* q2_h  = (_Float16*)alloc((size_t)MQ * D_ * 2);
    _Float16* src_h = (_Float16*)alloc((size_t)MV * D_ * 2);
    _Float16* val_h = (_Float16*)alloc((size_t)MV * D_ * 2);
    float*    off_f = (float*)   alloc((size_t)MQ * D_ * 4);
    float*    aw_f  = (float*)   alloc((size_t)MQ * 128 * 4);
    _Float16* ca_h  = (_Float16*)alloc((size_t)MQ * D_ * 2);
    float*    caout = (float*)   alloc((size_t)MQ * D_ * 4);
    float*    tgt2_f= (float*)   alloc((size_t)MQ * D_ * 4);
    _Float16* tgt2_h= (_Float16*)alloc((size_t)MQ * D_ * 2);
    _Float16* hid_h = (_Float16*)alloc((size_t)MQ * DFF_ * 2);
    float*    ffn_f = (float*)   alloc((size_t)MQ * D_ * 4);

    auto gblk = [](long long n) { return (unsigned)((n + 255) / 256); };
    auto gtile = [](int M, int N) { return dim3((unsigned)((N + 63) / 64),
                                                (unsigned)((M + 63) / 64)); };

    // weight transposes (f32 [K][N] -> f16 [N][K])
    k_transpose_w<<<gblk(256 * 256), 256, 0, stream>>>(w_q,  wt_q,  256, 256);
    k_transpose_w<<<gblk(256 * 256), 256, 0, stream>>>(w_k,  wt_k,  256, 256);
    k_transpose_w<<<gblk(256 * 256), 256, 0, stream>>>(w_v,  wt_v,  256, 256);
    k_transpose_w<<<gblk(256 * 256), 256, 0, stream>>>(w_o,  wt_o,  256, 256);
    k_transpose_w<<<gblk(256 * 256), 256, 0, stream>>>(w_vp, wt_vp, 256, 256);
    k_transpose_w<<<gblk(256 * 256), 256, 0, stream>>>(w_so, wt_so, 256, 256);
    k_transpose_w<<<gblk(256 * 128), 256, 0, stream>>>(w_aw, wt_aw, 256, 128);
    k_transpose_w<<<gblk(256 * 256), 256, 0, stream>>>(w_op, wt_op, 256, 256);
    k_transpose_w<<<gblk(256 * 1024), 256, 0, stream>>>(w_l1, wt_l1, 256, 1024);
    k_transpose_w<<<gblk(1024 * 256), 256, 0, stream>>>(w_l2, wt_l2, 1024, 256);

    long long nq = (long long)MQ * D_;
    k_addcast_f16<<<gblk(nq), 256, 0, stream>>>(tgt, qpos, qin_h, nq);
    k_cast_f16<<<gblk(nq), 256, 0, stream>>>(tgt, tgt_h, nq);

    // q/k/v projections -> f16
    k_gemm<<<gtile(MQ, D_), 256, 0, stream>>>(qin_h, wt_q, b_q, nullptr, qh, MQ, D_, D_, 0);
    k_gemm<<<gtile(MQ, D_), 256, 0, stream>>>(qin_h, wt_k, b_k, nullptr, kh, MQ, D_, D_, 0);
    k_gemm<<<gtile(MQ, D_), 256, 0, stream>>>(tgt_h, wt_v, b_v, nullptr, vh, MQ, D_, D_, 0);

    // flash self-attention
    k_flash<<<B_ * NH_ * 57, 32, 0, stream>>>(qh, kh, vh, sa_h);

    // output projection + LN2 (also prepares ca query = ln2_out + pos as f16)
    k_gemm<<<gtile(MQ, D_), 256, 0, stream>>>(sa_h, wt_o, b_o, attn_f, nullptr, MQ, D_, D_, 0);
    k_layernorm<<<MQ / 8, 256, 0, stream>>>(attn_f, tgt, ln2g, ln2b, qpos, tgt1_f, q2_h, MQ);

    // value projection over src
    long long nsrc = (long long)MV * D_;
    k_cast_f16<<<gblk(nsrc), 256, 0, stream>>>(src, src_h, nsrc);
    k_gemm<<<gtile(MV, D_), 256, 0, stream>>>(src_h, wt_vp, b_vp, nullptr, val_h, MV, D_, D_, 0);
    k_mask_value<<<gblk(nsrc), 256, 0, stream>>>(val_h, pmask, nsrc);

    // sampling offsets + attention weights
    k_gemm<<<gtile(MQ, D_), 256, 0, stream>>>(q2_h, wt_so, b_so, off_f, nullptr, MQ, D_, D_, 0);
    k_gemm<<<gtile(MQ, 128), 256, 0, stream>>>(q2_h, wt_aw, b_aw, aw_f, nullptr, MQ, 128, D_, 0);
    k_softmax16<<<gblk(MQ * NH_), 256, 0, stream>>>(aw_f, MQ * NH_);

    // deformable gather + out_proj + LN1
    k_deform<<<MQ, 256, 0, stream>>>(val_h, off_f, aw_f, refp, ca_h);
    k_gemm<<<gtile(MQ, D_), 256, 0, stream>>>(ca_h, wt_op, b_op, caout, nullptr, MQ, D_, D_, 0);
    k_layernorm<<<MQ / 8, 256, 0, stream>>>(caout, tgt1_f, ln1g, ln1b, nullptr, tgt2_f, tgt2_h, MQ);

    // FFN + LN3 -> d_out
    k_gemm<<<gtile(MQ, DFF_), 256, 0, stream>>>(tgt2_h, wt_l1, b_l1, nullptr, hid_h, MQ, DFF_, D_, 1);
    k_gemm<<<gtile(MQ, D_), 256, 0, stream>>>(hid_h, wt_l2, b_l2, ffn_f, nullptr, MQ, D_, DFF_, 0);
    k_layernorm<<<MQ / 8, 256, 0, stream>>>(ffn_f, tgt2_f, ln3g, ln3b, nullptr,
                                            (float*)d_out, nullptr, MQ);
}
